// GCN__34007551050523
// MI455X (gfx1250) — matrix operand order repaired and checked by reading, not compiled, not run on hardware
//
#include <hip/hip_runtime.h>
#include <hip/hip_bf16.h>

typedef __attribute__((ext_vector_type(4)))  float  v4f;
typedef __attribute__((ext_vector_type(8)))  float  v8f;
typedef __attribute__((ext_vector_type(16))) __bf16 v16bf;

#define N_NODES 50000
#define IN_CH   512
#define OUT_CH  256
#define N_EDGES 800000
#define M_TILES (N_NODES / 16)   // 3125, exact
#define K2      (IN_CH / 2)      // 256 packed-K rows
#define BSTRIDE 20               // LDS row stride (dwords): +4 rows -> +80 -> bank shift 16

// ---- bf16 split helpers ----------------------------------------------------
static __device__ __forceinline__ void bf_split(float v, __bf16& hi, __bf16& lo) {
    __bf16 h = (__bf16)v;            // RNE
    hi = h;
    lo = (__bf16)(v - (float)h);     // residual exact in f32, then RNE to bf16
}

static __device__ __forceinline__ v4f fma4(v4f a, float s, v4f c) {
    v4f r;
    r.x = fmaf(a.x, s, c.x); r.y = fmaf(a.y, s, c.y);
    r.z = fmaf(a.z, s, c.z); r.w = fmaf(a.w, s, c.w);
    return r;
}

// ---------------------------------------------------------------- degree ----
__global__ void gcn_deg_zero(int* __restrict__ deg) {
    int i = blockIdx.x * blockDim.x + threadIdx.x;
    if (i < N_NODES) deg[i] = 0;
}

__global__ void gcn_deg_count(const int* __restrict__ ei, int* __restrict__ deg) {
    int e = blockIdx.x * blockDim.x + threadIdx.x;
    if (e < N_EDGES) atomicAdd(&deg[ei[N_EDGES + e]], 1);   // dst row
}

__global__ void gcn_deg_inv(const int* __restrict__ deg, float* __restrict__ dinv) {
    int i = blockIdx.x * blockDim.x + threadIdx.x;
    if (i < N_NODES) dinv[i] = rsqrtf((float)deg[i] + 1.0f);  // +1 self loop
}

// ----------------------------------------------------- CSR build (scan) -----
// Exclusive prefix sum of deg -> rowstart, single 1024-thread block.
__global__ __launch_bounds__(1024) void gcn_scan(const int* __restrict__ deg,
                                                 int* __restrict__ rowstart,
                                                 int* __restrict__ cursor) {
    __shared__ int buf[1024];
    __shared__ int carry;
    const int tid = threadIdx.x;
    if (tid == 0) carry = 0;
    __syncthreads();
    for (int base = 0; base < N_NODES; base += 1024) {
        const int i = base + tid;
        int v = (i < N_NODES) ? deg[i] : 0;
        buf[tid] = v;
        __syncthreads();
        #pragma unroll
        for (int ofs = 1; ofs < 1024; ofs <<= 1) {
            int t = (tid >= ofs) ? buf[tid - ofs] : 0;
            __syncthreads();
            buf[tid] += t;
            __syncthreads();
        }
        const int excl = carry + buf[tid] - v;
        if (i < N_NODES) { rowstart[i] = excl; cursor[i] = excl; }
        __syncthreads();
        if (tid == 1023) carry += buf[1023];
        __syncthreads();
    }
}

__global__ void gcn_fill(const int* __restrict__ ei, int* __restrict__ cursor,
                         int* __restrict__ adj) {
    int e = blockIdx.x * blockDim.x + threadIdx.x;
    if (e < N_EDGES) {
        const int s = ei[e];
        const int d = ei[N_EDGES + e];
        adj[atomicAdd(&cursor[d], 1)] = s;
    }
}

// --------------------------------------------- W -> bf16 hi/lo K-pair pack --
// Whp/Wlp[k2*256 + n] = bf16(W[2k2][n]) | bf16(W[2k2+1][n]) << 16
__global__ void gcn_pack_w(const float* __restrict__ W,
                           unsigned* __restrict__ Whp, unsigned* __restrict__ Wlp) {
    int i = blockIdx.x * blockDim.x + threadIdx.x;   // over K2*OUT_CH
    if (i >= K2 * OUT_CH) return;
    const int k2 = i >> 8, n = i & (OUT_CH - 1);
    const float f0 = W[(2 * k2)     * OUT_CH + n];
    const float f1 = W[(2 * k2 + 1) * OUT_CH + n];
    __bf16 h0, l0, h1, l1;
    bf_split(f0, h0, l0);
    bf_split(f1, h1, l1);
    Whp[i] = (unsigned)__builtin_bit_cast(unsigned short, h0) |
             ((unsigned)__builtin_bit_cast(unsigned short, h1) << 16);
    Wlp[i] = (unsigned)__builtin_bit_cast(unsigned short, l0) |
             ((unsigned)__builtin_bit_cast(unsigned short, l1) << 16);
}

// ------------------------------------------------------------- WMMA GEMM ----
// h = x @ W via 3-term bf16 split on V_WMMA_F32_16X16X32_BF16 (f32 accum):
//   h ~= xh@Wh + xh@Wl + xl@Wh   (xl@Wl term ~2^-16, dropped)
// Block = 8 waves = 128x16 tile. W hi/lo slabs in LDS, row stride 20 dwords
// so the two 16-lane halves hit disjoint bank groups.
__global__ __launch_bounds__(256) void gcn_gemm_wmma(
    const float* __restrict__ x, const unsigned* __restrict__ Whp,
    const unsigned* __restrict__ Wlp, float* __restrict__ h) {
    __shared__ unsigned Bh[K2 * BSTRIDE];
    __shared__ unsigned Bl[K2 * BSTRIDE];

    const int tid = threadIdx.x;
    const int n0  = blockIdx.y * 16;

    for (int i = tid; i < K2 * 16; i += 256) {
        const int k2 = i >> 4, j = i & 15;
        Bh[k2 * BSTRIDE + j] = Whp[k2 * OUT_CH + n0 + j];
        Bl[k2 * BSTRIDE + j] = Wlp[k2 * OUT_CH + n0 + j];
    }
    __syncthreads();

    const int wave = tid >> 5;
    const int lane = tid & 31;
    const int mt   = blockIdx.x * 8 + wave;
    if (mt >= M_TILES) return;

    const int m0   = mt * 16;
    const int lm   = lane & 15;
    const int hi16 = lane >> 4;       // 0: K 0..7/16..23, 1: K 8..15/24..31
    const int koff = hi16 << 3;

    const float* xrow = x + (size_t)(m0 + lm) * IN_CH;

    v8f acc = {};
    #pragma unroll 2
    for (int k0 = 0; k0 < IN_CH; k0 += 32) {
        // ---- A operand: 16 fp32 -> bf16 hi/lo, ISA 16-bit A layout ----
        v4f f0 = *(const v4f*)(xrow + k0 + koff);
        v4f f1 = *(const v4f*)(xrow + k0 + koff + 4);
        v4f g0 = *(const v4f*)(xrow + k0 + 16 + koff);
        v4f g1 = *(const v4f*)(xrow + k0 + 16 + koff + 4);
        float fv[16] = { f0.x, f0.y, f0.z, f0.w, f1.x, f1.y, f1.z, f1.w,
                         g0.x, g0.y, g0.z, g0.w, g1.x, g1.y, g1.z, g1.w };
        v16bf ah, al;
        #pragma unroll
        for (int i = 0; i < 16; ++i) {
            __bf16 hh, ll;
            bf_split(fv[i], hh, ll);
            ah[i] = hh;
            al[i] = ll;
        }
        // ---- B operand: packed K-pair dwords from LDS ----
        union { unsigned u[8]; v16bf v; } bh, bl;
        const int k2b = (k0 >> 1) + (hi16 << 2);
        #pragma unroll
        for (int r = 0; r < 8; ++r) {
            bh.u[r] = Bh[(k2b + r + (r & 4)) * BSTRIDE + lm];
            bl.u[r] = Bl[(k2b + r + (r & 4)) * BSTRIDE + lm];
        }
        acc = __builtin_amdgcn_wmma_f32_16x16x32_bf16(false, ah, false, bh.v,
                                                      (short)0, acc, false, false);
        acc = __builtin_amdgcn_wmma_f32_16x16x32_bf16(false, ah, false, bl.v,
                                                      (short)0, acc, false, false);
        acc = __builtin_amdgcn_wmma_f32_16x16x32_bf16(false, al, false, bh.v,
                                                      (short)0, acc, false, false);
    }

    // C/D: VGPR r -> M = r (lanes 0-15) / r+8 (lanes 16-31), N = lane%16
    const int rofs = hi16 << 3;
    float* hout = h + (size_t)(m0 + rofs) * OUT_CH + n0 + lm;
    #pragma unroll
    for (int r = 0; r < 8; ++r) hout[(size_t)r * OUT_CH] = acc[r];
}

// ------------------------------------------------------- CSR gather ---------
// One wave per dst node. acc = sum_in-edges dinv[s]*h[s] + dinv[n]*h[n];
// out[n] = relu(dinv[n]*acc + b). No f32 atomics; reads/writes coalesced.
__global__ __launch_bounds__(256) void gcn_gather(
    const int* __restrict__ rowstart, const int* __restrict__ adj,
    const float* __restrict__ h, const float* __restrict__ dinv,
    const float* __restrict__ b, float* __restrict__ out) {
    const int lane = threadIdx.x & 31;
    const int n    = blockIdx.x * 8 + (threadIdx.x >> 5);
    if (n >= N_NODES) return;

    const int beg = rowstart[n];
    const int end = (n + 1 < N_NODES) ? rowstart[n + 1] : N_EDGES;
    const float dn = dinv[n];
    const int c0 = lane * 4;          // channels c0..c0+3 and 128+c0..+3

    v4f acc0 = {}, acc1 = {};
    for (int p = beg; p < end; ++p) {
        const int s   = adj[p];
        const float w = dinv[s];
        const float* hs = h + (size_t)s * OUT_CH;
        v4f a0 = *(const v4f*)(hs + c0);
        v4f a1 = *(const v4f*)(hs + 128 + c0);
        acc0 = fma4(a0, w, acc0);
        acc1 = fma4(a1, w, acc1);
    }
    // self loop
    const float* hn = h + (size_t)n * OUT_CH;
    acc0 = fma4(*(const v4f*)(hn + c0),       dn, acc0);
    acc1 = fma4(*(const v4f*)(hn + 128 + c0), dn, acc1);

    v4f b0 = *(const v4f*)(b + c0);
    v4f b1 = *(const v4f*)(b + 128 + c0);
    v4f o0 = fma4(acc0, dn, b0);
    v4f o1 = fma4(acc1, dn, b1);
    o0.x = fmaxf(o0.x, 0.f); o0.y = fmaxf(o0.y, 0.f);
    o0.z = fmaxf(o0.z, 0.f); o0.w = fmaxf(o0.w, 0.f);
    o1.x = fmaxf(o1.x, 0.f); o1.y = fmaxf(o1.y, 0.f);
    o1.z = fmaxf(o1.z, 0.f); o1.w = fmaxf(o1.w, 0.f);

    float* on = out + (size_t)n * OUT_CH;
    *(v4f*)(on + c0)       = o0;
    *(v4f*)(on + 128 + c0) = o1;
}

// ------------------------------------------------------------ launcher ------
extern "C" void kernel_launch(void* const* d_in, const int* in_sizes, int n_in,
                              void* d_out, int out_size, void* d_ws, size_t ws_size,
                              hipStream_t stream) {
    const float* x  = (const float*)d_in[0];   // [50000, 512]
    const int*   ei = (const int*)  d_in[1];   // [2, 800000]
    const float* W  = (const float*)d_in[2];   // [512, 256]
    const float* b  = (const float*)d_in[3];   // [256]
    float* out = (float*)d_out;                // [50000, 256]

    char* ws = (char*)d_ws;
    float*    h        = (float*)    ws;                    // 51,200,000 B
    int*      deg      = (int*)     (ws + 51200000);        //    200,000 B
    float*    dinv     = (float*)   (ws + 51400000);        //    200,000 B
    int*      rowstart = (int*)     (ws + 51600000);        //    200,000 B
    int*      cursor   = (int*)     (ws + 51800000);        //    200,000 B
    int*      adj      = (int*)     (ws + 52000000);        //  3,200,000 B
    unsigned* Whp      = (unsigned*)(ws + 55200000);        //    262,144 B
    unsigned* Wlp      = (unsigned*)(ws + 55462144);        //    262,144 B

    // degree + symmetric norm
    gcn_deg_zero <<<dim3((N_NODES + 255) / 256), dim3(256), 0, stream>>>(deg);
    gcn_deg_count<<<dim3((N_EDGES + 255) / 256), dim3(256), 0, stream>>>(ei, deg);
    gcn_deg_inv  <<<dim3((N_NODES + 255) / 256), dim3(256), 0, stream>>>(deg, dinv);

    // CSR build: scan + fill (counting sort by dst)
    gcn_scan<<<dim3(1), dim3(1024), 0, stream>>>(deg, rowstart, cursor);
    gcn_fill<<<dim3((N_EDGES + 255) / 256), dim3(256), 0, stream>>>(ei, cursor, adj);

    // W bf16 hi/lo pack + GEMM h = x @ W
    gcn_pack_w<<<dim3((K2 * OUT_CH) / 256), dim3(256), 0, stream>>>(W, Whp, Wlp);
    gcn_gemm_wmma<<<dim3((M_TILES + 7) / 8, OUT_CH / 16), dim3(256), 0, stream>>>(x, Whp, Wlp, h);

    // fused gather + self-loop + bias + relu
    gcn_gather<<<dim3((N_NODES + 7) / 8), dim3(256), 0, stream>>>(rowstart, adj, h, dinv, b, out);
}
